// GDspaMamba_55937654063133
// MI455X (gfx1250) — matrix operand order repaired and checked by reading, not compiled, not run on hardware
//
#include <hip/hip_runtime.h>

#define DEVFN __device__ __forceinline__

typedef __attribute__((ext_vector_type(16))) _Float16 v16h;
typedef __attribute__((ext_vector_type(8)))  _Float16 v8h;
typedef __attribute__((ext_vector_type(8)))  float    v8f;

namespace cfg {
constexpr int Cc   = 128;
constexpr int DI   = 256;
constexpr int Nst  = 16;
constexpr int R    = 8;
constexpr int Kdir = 4;
constexpr int Bb   = 4;
constexpr int H    = 48, W = 48;
constexpr int L    = H * W;        // 2304
constexpr int BL   = Bb * L;       // 9216
constexpr int G    = 16;           // scan chunks
constexpr int Lc   = L / G;        // 144
constexpr int XD   = R + 2 * Nst;  // 40
}

// ---------------- WMMA fragment helpers (CDNA5 16x16x32 f16 layouts) -------
// A (16x32, row-major source, lda halves): lane m=lane&15, hk=lane>>4
//   half i -> K = k0 + 16*hk + i  => one 32B contiguous load
DEVFN v16h load_a_frag(const _Float16* rowk0, int hk) {
  return *(const v16h*)(rowk0 + 16 * hk);
}
// B (32x16) from weight stored row-major [N][K]: lane n=lane&15, hk=lane>>4
//   half i -> K = k0 + 16*hk + i  => one 32B contiguous load
DEVFN v16h load_b_frag(const _Float16* wrowk0, int hk) {
  return *(const v16h*)(wrowk0 + 16 * hk);
}

// scan-direction l -> pixel index (h*W+w)
DEVFN int dir_pix(int l, int k) {
  int ll = (k >= 2) ? (cfg::L - 1 - l) : l;
  if (k & 1) { int h = ll % cfg::H; int w = ll / cfg::H; return h * cfg::W + w; }
  return ll;
}

// ---------------- K0: weight f32 -> f16 conversion (x_proj padded to 48) ---
__global__ void k_convert(const float* __restrict__ w1, const float* __restrict__ xpw,
                          const float* __restrict__ wo,
                          _Float16* __restrict__ w1h, _Float16* __restrict__ xpwh,
                          _Float16* __restrict__ woh) {
  int t = blockIdx.x * 256 + threadIdx.x;
  const int S1 = 512 * 128;
  const int S2 = 4 * 48 * 256;
  const int S3 = 128 * 256;
  if (t < S1) { w1h[t] = (_Float16)w1[t]; return; }
  t -= S1;
  if (t < S2) {
    int d = t % 256, rem = t / 256;
    int c = rem % 48, k = rem / 48;
    float v = (c < cfg::XD) ? xpw[((k * cfg::XD + c) * 256) + d] : 0.f;
    xpwh[t] = (_Float16)v;
    return;
  }
  t -= S2;
  if (t < S3) woh[t] = (_Float16)wo[t];
}

// ---------------- K1: layernorm over C, emit f16 ---------------------------
__global__ void k_ln1(const float* __restrict__ x, const float* __restrict__ g,
                      const float* __restrict__ bia, _Float16* __restrict__ xn) {
  const int lane = threadIdx.x & 31;
  const int wave = threadIdx.x >> 5;
  const int p = blockIdx.x * 8 + wave;          // pixel 0..BL-1
  const int b = p / cfg::L, l = p % cfg::L;
  const float* px = x + (size_t)b * cfg::Cc * cfg::L + l;
  float v[4];
  float s = 0.f, s2 = 0.f;
#pragma unroll
  for (int j = 0; j < 4; ++j) {
    int c = lane + 32 * j;
    v[j] = px[(size_t)c * cfg::L];
    s += v[j]; s2 += v[j] * v[j];
  }
#pragma unroll
  for (int off = 16; off; off >>= 1) { s += __shfl_xor(s, off, 32); s2 += __shfl_xor(s2, off, 32); }
  const float m  = s  * (1.f / cfg::Cc);
  const float vr = s2 * (1.f / cfg::Cc) - m * m;
  const float rs = rsqrtf(vr + 1e-5f);
#pragma unroll
  for (int j = 0; j < 4; ++j) {
    int c = lane + 32 * j;
    xn[(size_t)p * cfg::Cc + c] = (_Float16)((v[j] - m) * rs * g[c] + bia[c]);
  }
}

// ---------------- K2: in_proj GEMM (9216x512x128) via WMMA -----------------
__global__ void k_gemm_inproj(const _Float16* __restrict__ xn, const _Float16* __restrict__ w1,
                              float* __restrict__ xz) {
  const int lane = threadIdx.x & 31;
  const int wave = threadIdx.x >> 5;
  const int tile = blockIdx.x * 8 + wave;
  const int MT = cfg::BL / 16;                  // 576
  const int mt = tile % MT, nt = tile / MT;     // nt 0..31
  const int m0 = mt * 16, n0 = nt * 16;
  const int hk = lane >> 4, lr = lane & 15;
  v8f acc = {};
#pragma unroll
  for (int kk = 0; kk < 4; ++kk) {
    const int k0 = kk * 32;
    v16h a = load_a_frag(xn + (size_t)(m0 + lr) * 128 + k0, hk);
    v16h b = load_b_frag(w1 + (size_t)(n0 + lr) * 128 + k0, hk);
    acc = __builtin_amdgcn_wmma_f32_16x16x32_f16(false, a, false, b, (short)0, acc, false, false);
  }
#pragma unroll
  for (int j = 0; j < 8; ++j) {
    int m = m0 + j + 8 * hk;
    xz[(size_t)m * 512 + n0 + lr] = acc[j];
  }
}

// ---------------- K3: depthwise 3x3 conv + bias + SiLU ---------------------
__global__ void k_conv(const float* __restrict__ xz, const float* __restrict__ cw,
                       const float* __restrict__ cb,
                       _Float16* __restrict__ xch, float* __restrict__ xcf) {
  const int d = threadIdx.x;                    // 0..255
  const int p = blockIdx.x;                     // 0..BL-1
  const int b = p / cfg::L, l = p % cfg::L;
  const int h = l / cfg::W, w = l % cfg::W;
  float acc = cb[d];
#pragma unroll
  for (int dh = -1; dh <= 1; ++dh)
#pragma unroll
    for (int dw = -1; dw <= 1; ++dw) {
      int hh = h + dh, ww = w + dw;
      if (hh >= 0 && hh < cfg::H && ww >= 0 && ww < cfg::W) {
        float xv = xz[((size_t)b * cfg::L + hh * cfg::W + ww) * 512 + d];
        acc += xv * cw[d * 9 + (dh + 1) * 3 + (dw + 1)];
      }
    }
  float sv = acc / (1.f + __expf(-acc));
  xcf[(size_t)p * cfg::DI + d] = sv;
  xch[(size_t)p * cfg::DI + d] = (_Float16)sv;
}

// ---------------- K4: x_proj GEMM per (b,k), direction folded into gather --
__global__ void k_gemm_xproj(const _Float16* __restrict__ xch, const _Float16* __restrict__ xpw,
                             float* __restrict__ xdbl) {
  const int lane = threadIdx.x & 31;
  const int wave = threadIdx.x >> 5;
  int tile = blockIdx.x * 8 + wave;
  const int LT = cfg::L / 16;                   // 144
  const int ct = tile % 3; tile /= 3;
  const int lt = tile % LT; tile /= LT;
  const int k  = tile % cfg::Kdir; tile /= cfg::Kdir;
  const int b  = tile;
  const int m0 = lt * 16, c0 = ct * 16;
  const int hk = lane >> 4, lr = lane & 15;
  const int pix = dir_pix(m0 + lr, k);
  const _Float16* arow = xch + ((size_t)b * cfg::L + pix) * cfg::DI;
  const _Float16* brow = xpw + ((size_t)k * 48 + c0 + lr) * cfg::DI;
  v8f acc = {};
#pragma unroll
  for (int kk = 0; kk < 8; ++kk) {
    v16h a  = load_a_frag(arow + kk * 32, hk);
    v16h bf = load_b_frag(brow + kk * 32, hk);
    acc = __builtin_amdgcn_wmma_f32_16x16x32_f16(false, a, false, bf, (short)0, acc, false, false);
  }
  const int c = c0 + lr;
  if (c < cfg::XD) {
#pragma unroll
    for (int j = 0; j < 8; ++j) {
      int m = m0 + j + 8 * hk;
      xdbl[(((size_t)b * cfg::Kdir + k) * cfg::L + m) * cfg::XD + c] = acc[j];
    }
  }
}

// ---------------- K5: delta = softplus(dts @ dt_w^T + dt_b) (R=8, VALU) ----
__global__ void k_delta(const float* __restrict__ xdbl, const float* __restrict__ dtw,
                        const float* __restrict__ dtb, float* __restrict__ delta) {
  const int d = threadIdx.x;
  const int bkl = blockIdx.x;                   // ((b*K+k)*L + l)
  const int k = (bkl / cfg::L) % cfg::Kdir;
  const float* dts  = xdbl + (size_t)bkl * cfg::XD;
  const float* wrow = dtw + ((size_t)k * cfg::DI + d) * cfg::R;
  float acc = dtb[k * cfg::DI + d];
#pragma unroll
  for (int r = 0; r < cfg::R; ++r) acc = fmaf(dts[r], wrow[r], acc);
  delta[(size_t)bkl * cfg::DI + d] = (acc > 20.f) ? acc : log1pf(__expf(acc));
}

// ---------------- K6a: chunked scan pass 1: per-chunk (prod a, partial h) --
__global__ void k_scan1(const float* __restrict__ xcf, const float* __restrict__ delta,
                        const float* __restrict__ xdbl, const float* __restrict__ alog,
                        float* __restrict__ cP, float* __restrict__ cR) {
  const int d = threadIdx.x;
  int id = blockIdx.x;
  const int g = id % cfg::G; id /= cfg::G;
  const int k = id % cfg::Kdir; id /= cfg::Kdir;
  const int b = id;
  float A[16];
#pragma unroll
  for (int n = 0; n < 16; ++n) A[n] = -__expf(alog[((size_t)k * cfg::DI + d) * 16 + n]);
  float P[16], Rr[16];
#pragma unroll
  for (int n = 0; n < 16; ++n) { P[n] = 1.f; Rr[n] = 0.f; }
  const size_t bk = (size_t)b * cfg::Kdir + k;
  for (int l = g * cfg::Lc; l < (g + 1) * cfg::Lc; ++l) {
    const int pix = dir_pix(l, k);
    const float u  = xcf[((size_t)b * cfg::L + pix) * cfg::DI + d];
    const float dt = delta[(bk * cfg::L + l) * cfg::DI + d];
    const float du = dt * u;
    const float* bc = xdbl + (bk * cfg::L + l) * cfg::XD + cfg::R;
#pragma unroll
    for (int n = 0; n < 16; ++n) {
      float an = __expf(dt * A[n]);
      Rr[n] = fmaf(an, Rr[n], du * bc[n]);
      P[n] *= an;
    }
  }
  const size_t o = ((bk * cfg::G + g) * cfg::DI + d) * 16;
#pragma unroll
  for (int n = 0; n < 16; ++n) { cP[o + n] = P[n]; cR[o + n] = Rr[n]; }
}

// ---------------- K6b: stitch chunk states (16 steps) ----------------------
__global__ void k_scan2(const float* __restrict__ cP, const float* __restrict__ cR,
                        float* __restrict__ hst) {
  const int d = threadIdx.x;
  const size_t bk = blockIdx.x;
  float h[16];
#pragma unroll
  for (int n = 0; n < 16; ++n) h[n] = 0.f;
  for (int g = 0; g < cfg::G; ++g) {
    const size_t o = ((bk * cfg::G + g) * cfg::DI + d) * 16;
#pragma unroll
    for (int n = 0; n < 16; ++n) {
      hst[o + n] = h[n];
      h[n] = fmaf(cP[o + n], h[n], cR[o + n]);
    }
  }
}

// ---------------- K6c: pass 3, replay with h0 and emit y = C.h + D*u -------
__global__ void k_scan3(const float* __restrict__ xcf, const float* __restrict__ delta,
                        const float* __restrict__ xdbl, const float* __restrict__ alog,
                        const float* __restrict__ Ds, const float* __restrict__ hst,
                        float* __restrict__ ys) {
  const int d = threadIdx.x;
  int id = blockIdx.x;
  const int g = id % cfg::G; id /= cfg::G;
  const int k = id % cfg::Kdir; id /= cfg::Kdir;
  const int b = id;
  float A[16];
#pragma unroll
  for (int n = 0; n < 16; ++n) A[n] = -__expf(alog[((size_t)k * cfg::DI + d) * 16 + n]);
  const size_t bk = (size_t)b * cfg::Kdir + k;
  float h[16];
  {
    const size_t o = ((bk * cfg::G + g) * cfg::DI + d) * 16;
#pragma unroll
    for (int n = 0; n < 16; ++n) h[n] = hst[o + n];
  }
  const float dsv = Ds[k * cfg::DI + d];
  for (int l = g * cfg::Lc; l < (g + 1) * cfg::Lc; ++l) {
    const int pix = dir_pix(l, k);
    const float u  = xcf[((size_t)b * cfg::L + pix) * cfg::DI + d];
    const float dt = delta[(bk * cfg::L + l) * cfg::DI + d];
    const float du = dt * u;
    const float* bc = xdbl + (bk * cfg::L + l) * cfg::XD + cfg::R;
    const float* cc = bc + 16;
    float y = dsv * u;
#pragma unroll
    for (int n = 0; n < 16; ++n) {
      float an = __expf(dt * A[n]);
      h[n] = fmaf(an, h[n], du * bc[n]);
      y = fmaf(h[n], cc[n], y);
    }
    ys[(bk * cfg::L + l) * cfg::DI + d] = y;
  }
}

// ---------------- K7: combine 4 directions + LN(DI) + silu(z) gate ---------
__global__ void k_combine(const float* __restrict__ ys, const float* __restrict__ xz,
                          const float* __restrict__ og, const float* __restrict__ ob,
                          _Float16* __restrict__ yg) {
  const int lane = threadIdx.x & 31;
  const int wave = threadIdx.x >> 5;
  const int p = blockIdx.x * 8 + wave;
  const int b = p / cfg::L, l = p % cfg::L;
  const int h = l / cfg::W, w = l % cfg::W;
  const int l1 = w * cfg::H + h;
  const size_t base = (size_t)b * cfg::Kdir;
  float v[8];
  float s = 0.f, s2 = 0.f;
#pragma unroll
  for (int j = 0; j < 8; ++j) {
    int d = j * 32 + lane;
    float t = ys[((base + 0) * cfg::L + l) * cfg::DI + d]
            + ys[((base + 2) * cfg::L + (cfg::L - 1 - l)) * cfg::DI + d]
            + ys[((base + 1) * cfg::L + l1) * cfg::DI + d]
            + ys[((base + 3) * cfg::L + (cfg::L - 1 - l1)) * cfg::DI + d];
    v[j] = t; s += t; s2 += t * t;
  }
#pragma unroll
  for (int off = 16; off; off >>= 1) { s += __shfl_xor(s, off, 32); s2 += __shfl_xor(s2, off, 32); }
  const float m  = s  * (1.f / cfg::DI);
  const float vr = s2 * (1.f / cfg::DI) - m * m;
  const float rs = rsqrtf(vr + 1e-5f);
#pragma unroll
  for (int j = 0; j < 8; ++j) {
    int d = j * 32 + lane;
    float zv  = xz[(size_t)p * 512 + 256 + d];
    float sz  = zv / (1.f + __expf(-zv));
    float lnv = (v[j] - m) * rs * og[d] + ob[d];
    yg[(size_t)p * cfg::DI + d] = (_Float16)(lnv * sz);
  }
}

// ---------------- K8: out_proj GEMM + residual, write (B,C,H,W) ------------
__global__ void k_gemm_outproj(const _Float16* __restrict__ yg, const _Float16* __restrict__ wo,
                               const float* __restrict__ x, float* __restrict__ out) {
  const int lane = threadIdx.x & 31;
  const int wave = threadIdx.x >> 5;
  const int tile = blockIdx.x * 8 + wave;
  const int NT = cfg::Cc / 16;                  // 8
  const int nt = tile % NT, mt = tile / NT;
  const int m0 = mt * 16, n0 = nt * 16;
  const int hk = lane >> 4, lr = lane & 15;
  v8f acc = {};
#pragma unroll
  for (int kk = 0; kk < 8; ++kk) {
    const int k0 = kk * 32;
    v16h a = load_a_frag(yg + (size_t)(m0 + lr) * cfg::DI + k0, hk);
    v16h b = load_b_frag(wo + (size_t)(n0 + lr) * cfg::DI + k0, hk);
    acc = __builtin_amdgcn_wmma_f32_16x16x32_f16(false, a, false, b, (short)0, acc, false, false);
  }
  const int c = n0 + lr;
#pragma unroll
  for (int j = 0; j < 8; ++j) {
    int m = m0 + j + 8 * hk;
    int b_ = m / cfg::L, l = m % cfg::L;
    size_t o = ((size_t)b_ * cfg::Cc + c) * cfg::L + l;
    out[o] = x[o] + acc[j];
  }
}

// ---------------- host launcher --------------------------------------------
extern "C" void kernel_launch(void* const* d_in, const int* in_sizes, int n_in,
                              void* d_out, int out_size, void* d_ws, size_t ws_size,
                              hipStream_t stream) {
  (void)in_sizes; (void)n_in; (void)out_size; (void)ws_size;
  using namespace cfg;
  const float* x    = (const float*)d_in[0];
  const float* ln1g = (const float*)d_in[1];
  const float* ln1b = (const float*)d_in[2];
  const float* w1   = (const float*)d_in[3];
  const float* cw   = (const float*)d_in[4];
  const float* cb   = (const float*)d_in[5];
  const float* xpw  = (const float*)d_in[6];
  const float* dtw  = (const float*)d_in[7];
  const float* dtb  = (const float*)d_in[8];
  const float* alog = (const float*)d_in[9];
  const float* Ds   = (const float*)d_in[10];
  const float* og   = (const float*)d_in[11];
  const float* ob   = (const float*)d_in[12];
  const float* wo   = (const float*)d_in[13];
  float* out = (float*)d_out;

  char* wsb = (char*)d_ws;
  size_t off = 0;
  auto alloc = [&](size_t bytes) -> void* {
    void* p = wsb + off;
    off = (off + bytes + 255) & ~(size_t)255;
    return p;
  };
  _Float16* xn_h = (_Float16*)alloc((size_t)BL * Cc * 2);
  _Float16* w1h  = (_Float16*)alloc((size_t)512 * 128 * 2);
  _Float16* xpwh = (_Float16*)alloc((size_t)4 * 48 * 256 * 2);
  _Float16* woh  = (_Float16*)alloc((size_t)128 * 256 * 2);
  float*    xzf  = (float*)alloc((size_t)BL * 512 * 4);
  _Float16* xch  = (_Float16*)alloc((size_t)BL * DI * 2);
  float*    xcf  = (float*)alloc((size_t)BL * DI * 4);
  float*    xdbl = (float*)alloc((size_t)Bb * Kdir * L * XD * 4);
  float*    dlt  = (float*)alloc((size_t)Bb * Kdir * L * DI * 4);
  float*    cP   = (float*)alloc((size_t)Bb * Kdir * G * DI * 16 * 4);
  float*    cR   = (float*)alloc((size_t)Bb * Kdir * G * DI * 16 * 4);
  float*    hst  = (float*)alloc((size_t)Bb * Kdir * G * DI * 16 * 4);
  float*    ysb  = (float*)alloc((size_t)Bb * Kdir * L * DI * 4);
  _Float16* yg   = (_Float16*)alloc((size_t)BL * DI * 2);

  k_convert<<<576, 256, 0, stream>>>(w1, xpw, wo, w1h, xpwh, woh);
  k_ln1<<<BL / 8, 256, 0, stream>>>(x, ln1g, ln1b, xn_h);
  k_gemm_inproj<<<(BL / 16) * (512 / 16) / 8, 256, 0, stream>>>(xn_h, w1h, xzf);
  k_conv<<<BL, 256, 0, stream>>>(xzf, cw, cb, xch, xcf);
  k_gemm_xproj<<<(Bb * Kdir * (L / 16) * 3) / 8, 256, 0, stream>>>(xch, xpwh, xdbl);
  k_delta<<<Bb * Kdir * L, 256, 0, stream>>>(xdbl, dtw, dtb, dlt);
  k_scan1<<<Bb * Kdir * G, 256, 0, stream>>>(xcf, dlt, xdbl, alog, cP, cR);
  k_scan2<<<Bb * Kdir, 256, 0, stream>>>(cP, cR, hst);
  k_scan3<<<Bb * Kdir * G, 256, 0, stream>>>(xcf, dlt, xdbl, alog, Ds, hst, ysb);
  k_combine<<<BL / 8, 256, 0, stream>>>(ysb, xzf, og, ob, yg);
  k_gemm_outproj<<<(BL / 16) * (Cc / 16) / 8, 256, 0, stream>>>(yg, woh, x, out);
}
